// Retriever_33354716021177
// MI455X (gfx1250) — matrix operand order repaired and checked by reading, not compiled, not run on hardware
//
#include <hip/hip_runtime.h>
#include <cstddef>

#define B_ROWS  1024
#define DDIM    512
#define NE_CNT  200000
#define LDIM    4
#define KSEL    33      // top (32+1)
#define TOPK    32
#define CHUNK   2048    // evidence per K1 block
#define NCHUNKS 98      // ceil(200000 / 2048)
#define MTILE   64      // rows per K1 block
#define NSTEP   128     // evidence per N-iteration (8 waves * 16)
#define SCW     132     // padded LDS score-tile stride

typedef _Float16 v8h  __attribute__((ext_vector_type(8)));
typedef _Float16 v16h __attribute__((ext_vector_type(16)));
typedef float    v16f __attribute__((ext_vector_type(16)));
typedef float    v8f  __attribute__((ext_vector_type(8)));

union V16H { v16h v; v8h h[2]; };

// ---------------------------------------------------------------- K0: split A
// graph_feature (f32) -> hi = f16(x), lo = f16(x - f32(hi)).  2-term split gives
// ~2^-22 relative dot-product accuracy with 3 f16 WMMAs instead of 8x-slower f32 WMMA.
__global__ __launch_bounds__(256) void k0_split(const float* __restrict__ x,
                                                _Float16* __restrict__ hi,
                                                _Float16* __restrict__ lo) {
  int i = blockIdx.x * 256 + threadIdx.x;
  float v = x[i];
  _Float16 h = (_Float16)v;
  hi[i] = h;
  lo[i] = (_Float16)(v - (float)h);
}

// ------------------------------------------- K1: score GEMM + per-chunk top-33
// Block tile: 64 rows x 128 evidence. 8 waves; wave w owns evidence cols [16w,16w+16)
// and loops 4 M-subtiles -> each converted B tile feeds 12 v_wmma_f32_16x16x32_f16.
__global__ __launch_bounds__(256) void k1_score_topk(
    const float* __restrict__ evid,
    const _Float16* __restrict__ Ahi,
    const _Float16* __restrict__ Alo,
    float* __restrict__ cand_val,
    int* __restrict__ cand_idx)
{
  __shared__ float sc[MTILE * SCW];     // 64x128 score tile (padded)
  __shared__ float lV[MTILE * KSEL];    // per-row running top-33 values
  __shared__ int   lI[MTILE * KSEL];    // per-row running top-33 indices

  const int tid   = threadIdx.x;
  const int wave  = tid >> 5;
  const int lane  = tid & 31;
  const int lhalf = lane >> 4;          // 0|1: lane group
  const int lrow  = lane & 15;          // row (A) / col (B) within 16x16 tile
  const int row0  = blockIdx.x * MTILE;
  const int ech0  = blockIdx.y * CHUNK;

  for (int i = tid; i < MTILE * KSEL; i += 256) { lV[i] = -1e30f; lI[i] = 0; }
  __syncthreads();

  const bool owner = (tid & 3) == 0;    // 64 owner threads spread over all waves
  const int  orow  = tid >> 2;
  float myMin = -1e30f;                 // register-cached 33rd-best threshold
  int   myPos = 0;

  for (int ns = 0; ns < CHUNK / NSTEP; ++ns) {
    const int e0 = ech0 + ns * NSTEP + wave * 16;
    int er = e0 + lrow;
    if (er >= NE_CNT) er = NE_CNT - 1;  // clamp; masked at scan time

    v8f acc[4] = {};

    for (int k0 = 0; k0 < DDIM; k0 += 32) {
      // B layout (32x16 f16): lane = col, K = 16*lhalf + j (contiguous) -> one 64B load
      const int kkB = k0 + 16 * lhalf;
      v16f bf = *(const v16f*)(evid + (size_t)er * DDIM + kkB);
      V16H Bh, Bl;
#pragma unroll
      for (int j = 0; j < 16; ++j) {
        float x = bf[j];
        _Float16 h = (_Float16)x;
        Bh.v[j] = h;
        Bl.v[j] = (_Float16)(x - (float)h);
      }
      // A layout (16x32 f16): lane = row, K = kkA + {0..15} (contiguous)
      const int kkA = k0 + 16 * lhalf;
#pragma unroll
      for (int m = 0; m < 4; ++m) {
        const size_t abase = (size_t)(row0 + m * 16 + lrow) * DDIM + kkA;
        V16H Av, Lv;
        Av.h[0] = *(const v8h*)(Ahi + abase);
        Av.h[1] = *(const v8h*)(Ahi + abase + 8);
        Lv.h[0] = *(const v8h*)(Alo + abase);
        Lv.h[1] = *(const v8h*)(Alo + abase + 8);
        acc[m] = __builtin_amdgcn_wmma_f32_16x16x32_f16(false, Av.v, false, Bh.v,
                                                        (short)0, acc[m], false, false);
        acc[m] = __builtin_amdgcn_wmma_f32_16x16x32_f16(false, Av.v, false, Bl.v,
                                                        (short)0, acc[m], false, false);
        acc[m] = __builtin_amdgcn_wmma_f32_16x16x32_f16(false, Lv.v, false, Bh.v,
                                                        (short)0, acc[m], false, false);
      }
    }

    // C layout: lane l, vgpr v -> (M = 8*lhalf + v, N = lrow) within 16x16 tile
#pragma unroll
    for (int m = 0; m < 4; ++m)
#pragma unroll
      for (int v = 0; v < 8; ++v)
        sc[(m * 16 + 8 * lhalf + v) * SCW + wave * 16 + lrow] = acc[m][v];
    __syncthreads();

    if (owner) {
      const int gbase = ech0 + ns * NSTEP;
      for (int col = 0; col < NSTEP; ++col) {
        int gi = gbase + col;
        if (gi >= NE_CNT) break;
        float v = sc[orow * SCW + col];
        if (v > myMin) {                // rare once list warms up
          lV[orow * KSEL + myPos] = v;
          lI[orow * KSEL + myPos] = gi;
          float mn = 1e30f; int mp = 0;
          for (int j = 0; j < KSEL; ++j) {
            float t = lV[orow * KSEL + j];
            if (t < mn) { mn = t; mp = j; }
          }
          myMin = mn; myPos = mp;
        }
      }
    }
    __syncthreads();
  }

  if (owner) {
    size_t base = ((size_t)(row0 + orow) * NCHUNKS + blockIdx.y) * KSEL;
    for (int j = 0; j < KSEL; ++j) {
      cand_val[base + j] = lV[orow * KSEL + j];
      cand_idx[base + j] = lI[orow * KSEL + j];
    }
  }
}

// --------------------------------- K2: merge chunk candidates, sqrt + softmax
__global__ __launch_bounds__(256) void k2_merge(const float* __restrict__ cand_val,
                                                const int* __restrict__ cand_idx,
                                                int* __restrict__ widx,
                                                float* __restrict__ wval)
{
  const int NC = NCHUNKS * KSEL;        // 3234 candidates per row
  __shared__ float cv[NCHUNKS * KSEL];
  __shared__ int   ci[NCHUNKS * KSEL];
  __shared__ float redV[256];
  __shared__ int   redP[256];
  __shared__ float selV[KSEL];
  __shared__ int   selI[KSEL];
  __shared__ float wgt[TOPK];

  const int row = blockIdx.x;
  const int tid = threadIdx.x;

  for (int i = tid; i < NC; i += 256) {
    cv[i] = cand_val[(size_t)row * NC + i];
    ci[i] = cand_idx[(size_t)row * NC + i];
  }
  __syncthreads();

  for (int t = 0; t < KSEL; ++t) {      // 33 parallel argmax passes
    float bv = -1e30f; int bp = -1;
    for (int i = tid; i < NC; i += 256) {
      float v = cv[i];
      if (v > bv) { bv = v; bp = i; }
    }
    redV[tid] = bv; redP[tid] = bp;
    __syncthreads();
    for (int s = 128; s > 0; s >>= 1) {
      if (tid < s && redV[tid + s] > redV[tid]) {
        redV[tid] = redV[tid + s]; redP[tid] = redP[tid + s];
      }
      __syncthreads();
    }
    if (tid == 0) {
      int p = redP[0];
      selV[t] = redV[0];
      selI[t] = (p >= 0) ? ci[p] : 0;
      if (p >= 0) cv[p] = -1e30f;
    }
    __syncthreads();
  }

  if (tid == 0) {                       // drop rank-0; softmax over sqrt(rank 1..32)
    float m = sqrtf(fmaxf(selV[1], 0.f));
    float s = 0.f;
    for (int k = 0; k < TOPK; ++k) s += __expf(sqrtf(fmaxf(selV[k + 1], 0.f)) - m);
    float inv = 1.f / s;
    for (int k = 0; k < TOPK; ++k)
      wgt[k] = __expf(sqrtf(fmaxf(selV[k + 1], 0.f)) - m) * inv;
  }
  __syncthreads();
  if (tid < TOPK) {
    widx[row * TOPK + tid] = selI[tid + 1];
    wval[row * TOPK + tid] = wgt[tid];
  }
}

// ------------------------- K3: gather label embeddings + weighted accumulation
__global__ __launch_bounds__(256) void k3_gather(const float* __restrict__ label_emb,
                                                 const int* __restrict__ emb_label,
                                                 const int* __restrict__ widx,
                                                 const float* __restrict__ wval,
                                                 float* __restrict__ out)
{
  const int row = blockIdx.x;
  const int tid = threadIdx.x;
  __shared__ float w[TOPK];
  __shared__ int lab[LDIM][TOPK];

  if (tid < TOPK) {
    w[tid] = wval[row * TOPK + tid];
    int e = widx[row * TOPK + tid];
    for (int l = 0; l < LDIM; ++l)
      lab[l][tid] = emb_label[(size_t)l * NE_CNT + e];
  }
  __syncthreads();

  for (int d = tid; d < DDIM; d += 256) {
    float acc = 0.f;
#pragma unroll
    for (int l = 0; l < LDIM; ++l)
      for (int k = 0; k < TOPK; ++k)
        acc += label_emb[((size_t)(l << 6) + lab[l][k]) * DDIM + d] * w[k];
    out[(size_t)row * DDIM + d] = acc;
  }
}

// ------------------------------------------------------------------- launcher
extern "C" void kernel_launch(void* const* d_in, const int* in_sizes, int n_in,
                              void* d_out, int out_size, void* d_ws, size_t ws_size,
                              hipStream_t stream) {
  const float* gf = (const float*)d_in[0];   // (1024, 512)
  const float* ev = (const float*)d_in[1];   // (200000, 512)
  const float* le = (const float*)d_in[2];   // (4, 64, 512)
  const int*   el = (const int*)d_in[3];     // (4, 200000)
  float* out = (float*)d_out;                // (1024, 512)

  char* ws = (char*)d_ws;
  const size_t szA    = (size_t)B_ROWS * DDIM * sizeof(_Float16);        // 1 MB
  const size_t szCand = (size_t)B_ROWS * NCHUNKS * KSEL * sizeof(float); // ~13.2 MB
  const size_t szW    = (size_t)B_ROWS * TOPK * sizeof(float);           // 128 KB
  _Float16* Ahi = (_Float16*)(ws);
  _Float16* Alo = (_Float16*)(ws + szA);
  float* cand_val = (float*)(ws + 2 * szA);
  int*   cand_idx = (int*)  (ws + 2 * szA + szCand);
  int*   widx     = (int*)  (ws + 2 * szA + 2 * szCand);
  float* wval     = (float*)(ws + 2 * szA + 2 * szCand + szW);

  k0_split<<<(B_ROWS * DDIM) / 256, 256, 0, stream>>>(gf, Ahi, Alo);
  k1_score_topk<<<dim3(B_ROWS / MTILE, NCHUNKS), 256, 0, stream>>>(ev, Ahi, Alo,
                                                                   cand_val, cand_idx);
  k2_merge<<<B_ROWS, 256, 0, stream>>>(cand_val, cand_idx, widx, wval);
  k3_gather<<<B_ROWS, 256, 0, stream>>>(le, el, widx, wval, out);
}
